// SpikeDrivenSelfAttention_79379585565554
// MI455X (gfx1250) — 16-bit and 8-bit matrix operand order repaired and checked by reading, compiled, not run on hardware
//
#include <hip/hip_runtime.h>
#include <hip/hip_bf16.h>


typedef __attribute__((ext_vector_type(16))) __bf16 v16bf;
typedef __attribute__((ext_vector_type(8)))  float  v8f;
typedef __attribute__((ext_vector_type(8)))  int    v8i;
typedef __attribute__((ext_vector_type(4)))  int    v4i;
typedef __attribute__((ext_vector_type(2)))  int    v2i;

#define T_SEQ 2048
#define C_DIM 1024
#define NH    16
#define HD    64
#define BT    4096


__device__ __forceinline__ v16bf load_A_bf16(const __bf16* base, int row0,
                                             int k0, int lda, int lane) {
  int half = lane >> 4;
  int r    = row0 + (lane & 15);
  const __bf16* p = base + (size_t)r * lda + k0 + half * 8;
  union { v16bf v; v4i q[2]; } u;
  u.q[0] = *(const v4i*)(p);
  u.q[1] = *(const v4i*)(p + 16);
  return u.v;
}

__device__ __forceinline__ v16bf load_B_bf16(const __bf16* base, int n0,
                                             int k0, int ldb, int lane) {
  int half = lane >> 4;
  int n    = n0 + (lane & 15);
  const __bf16* p = base + (size_t)n * ldb + k0 + half * 8;
  union { v16bf v; v4i q[2]; } u;
  u.q[0] = *(const v4i*)(p);
  u.q[1] = *(const v4i*)(p + 16);
  return u.v;
}

__device__ __forceinline__ v8i load_A_iu8(const signed char* base, int row0,
                                          int ld, int lane) {
  int half = lane >> 4;
  int r    = row0 + (lane & 15);
  const signed char* p = base + (size_t)r * ld + half * 16;
  union { v8i v; v2i d[4]; } u;
  u.d[0] = *(const v2i*)(p);
  u.d[1] = *(const v2i*)(p + 8);
  u.d[2] = *(const v2i*)(p + 32);
  u.d[3] = *(const v2i*)(p + 40);
  return u.v;
}

__device__ __forceinline__ v8i load_B_iu8(const signed char* base, int n0,
                                          int lane) {
  int half = lane >> 4;
  int n    = n0 + (lane & 15);
  const signed char* p = base + (size_t)n * HD + half * 16;
  union { v8i v; v4i q[2]; } u;
  u.q[0] = *(const v4i*)(p);
  u.q[1] = *(const v4i*)(p + 32);
  return u.v;
}

__device__ __forceinline__ void async_copy16(unsigned lds_addr,
                                             const void* gaddr) {
  asm volatile("global_load_async_to_lds_b128 %0, %1, off"
               :: "v"(lds_addr), "v"((unsigned long long)(size_t)gaddr)
               : "memory");
}
__device__ __forceinline__ void wait_async0() {
  asm volatile("s_wait_asynccnt 0" ::: "memory");
}

__global__ void cvt_bf16_kernel(const float* __restrict__ in,
                                __bf16* __restrict__ out, int n) {
  int i = blockIdx.x * blockDim.x + threadIdx.x;
  if (i < n) out[i] = (__bf16)in[i];
}

__global__ __launch_bounds__(256) void qkv_kernel(
    const __bf16* __restrict__ xb,
    const __bf16* __restrict__ Wq, const __bf16* __restrict__ Wk,
    const __bf16* __restrict__ Wv,
    const float* __restrict__ bq, const float* __restrict__ bk,
    const float* __restrict__ bv,
    signed char* __restrict__ qs, signed char* __restrict__ ks,
    __bf16* __restrict__ vT) {
  int lane = threadIdx.x & 31;
  int wid  = blockIdx.x * 8 + (threadIdx.x >> 5);
  int mat  = wid / (256 * 16);
  int rem  = wid % (256 * 16);
  int m0   = (rem >> 4) * 16;
  int n0   = (rem & 15) * 64;
  const __bf16* W    = (mat == 0) ? Wq : (mat == 1) ? Wk : Wv;
  const float*  bias = (mat == 0) ? bq : (mat == 1) ? bk : bv;
  int half = lane >> 4, c = lane & 15;

  v8f acc[4];
#pragma unroll
  for (int t = 0; t < 4; ++t)
#pragma unroll
    for (int r = 0; r < 8; ++r) acc[t][r] = 0.f;

  for (int k0 = 0; k0 < C_DIM; k0 += 32) {
    __builtin_prefetch(xb + (size_t)(m0 + (lane & 15)) * C_DIM + k0 + 64, 0, 1);
    v16bf a = load_A_bf16(xb, m0, k0, C_DIM, lane);
#pragma unroll
    for (int t = 0; t < 4; ++t) {
      v16bf b = load_B_bf16(W, n0 + t * 16, k0, C_DIM, lane);
      acc[t] = __builtin_amdgcn_wmma_f32_16x16x32_bf16(
          false, a, false, b, (short)0, acc[t], false, false);
    }
  }

#pragma unroll
  for (int t = 0; t < 4; ++t) {
    int n = n0 + t * 16 + c;
    float bn = bias[n];
    int h = n >> 6, d = n & 63;
#pragma unroll
    for (int r = 0; r < 8; ++r) {
      int m   = m0 + half * 8 + r;
      int bb  = m >> 11;
      int tok = m & (T_SEQ - 1);
      float y = acc[t][r] + bn;
      if (mat == 2) {
        vT[(((size_t)(bb * NH + h)) * HD + d) * T_SEQ + tok] = (__bf16)y;
      } else {
        signed char sp = (y >= 1.0f) ? (signed char)1 : (signed char)0;
        signed char* dst = (mat == 0) ? qs : ks;
        dst[(((size_t)(bb * NH + h)) * T_SEQ + tok) * HD + d] = sp;
      }
    }
  }
}

__global__ __launch_bounds__(256) void attn_kernel(
    const signed char* __restrict__ qs, const signed char* __restrict__ ks,
    const __bf16* __restrict__ vT, __bf16* __restrict__ oh) {
  __shared__ __attribute__((aligned(16))) signed char ksb[2][32 * HD];
  __shared__ __attribute__((aligned(16))) __bf16      vtb[2][HD * 32];
  __shared__ __attribute__((aligned(16))) __bf16      pbuf[8][16 * 32];

  int tid  = threadIdx.x;
  int lane = tid & 31;
  int w    = tid >> 5;
  int bh   = blockIdx.x >> 4;
  int q0   = (blockIdx.x & 15) * 128 + w * 16;
  int half = lane >> 4, c = lane & 15;

  const signed char* qs_bh = qs + (size_t)bh * T_SEQ * HD;
  const signed char* ks_bh = ks + (size_t)bh * T_SEQ * HD;
  const __bf16*      vT_bh = vT + (size_t)bh * HD * T_SEQ;

  v8i qA = load_A_iu8(qs_bh, q0, HD, lane);

  float m8[8], l8[8];
  v8f acc[4];
#pragma unroll
  for (int r = 0; r < 8; ++r) { m8[r] = -1e30f; l8[r] = 0.f; }
#pragma unroll
  for (int t = 0; t < 4; ++t)
#pragma unroll
    for (int r = 0; r < 8; ++r) acc[t][r] = 0.f;

  auto stage = [&](int kc, signed char* ksd, __bf16* vtd) {
    if (tid < 128) {
      const signed char* g = ks_bh + (size_t)kc * HD + tid * 16;
      async_copy16((unsigned)(size_t)(ksd + tid * 16), g);
    }
    {
      int d = tid >> 2, seg = (tid & 3) * 16;
      const char* g = (const char*)(vT_bh + (size_t)d * T_SEQ + kc) + seg;
      async_copy16((unsigned)(size_t)((char*)(vtd + (size_t)d * 32) + seg), g);
    }
  };

  const float scale = 0.125f;
  v8i zero = {0, 0, 0, 0, 0, 0, 0, 0};

  stage(0, ksb[0], vtb[0]);
  wait_async0();
  __syncthreads();

  for (int kc = 0; kc < T_SEQ; kc += 32) {
    int cur = (kc >> 5) & 1;
    if (kc + 32 < T_SEQ)
      stage(kc + 32, ksb[cur ^ 1], vtb[cur ^ 1]);

    v8i b0 = load_B_iu8(ksb[cur], 0, lane);
    v8i b1 = load_B_iu8(ksb[cur], 16, lane);
    v8i s0 = __builtin_amdgcn_wmma_i32_16x16x64_iu8(false, qA, false, b0,
                                                    zero, false, false);
    v8i s1 = __builtin_amdgcn_wmma_i32_16x16x64_iu8(false, qA, false, b1,
                                                    zero, false, false);
    float p0[8], p1[8], sc[8];
#pragma unroll
    for (int r = 0; r < 8; ++r) {
      float f0 = (float)s0[r] * scale;
      float f1 = (float)s1[r] * scale;
      float rm = fmaxf(f0, f1);
#pragma unroll
      for (int off = 1; off < 16; off <<= 1)
        rm = fmaxf(rm, __shfl_xor(rm, off, 32));
      float nm = fmaxf(m8[r], rm);
      float s  = __expf(m8[r] - nm);
      float e0 = __expf(f0 - nm);
      float e1 = __expf(f1 - nm);
      float rs = e0 + e1;
#pragma unroll
      for (int off = 1; off < 16; off <<= 1) rs += __shfl_xor(rs, off, 32);
      l8[r] = l8[r] * s + rs;
      m8[r] = nm;
      sc[r] = s;
      p0[r] = e0; p1[r] = e1;
    }
#pragma unroll
    for (int t = 0; t < 4; ++t)
#pragma unroll
      for (int r = 0; r < 8; ++r) acc[t][r] *= sc[r];

#pragma unroll
    for (int r = 0; r < 8; ++r) {
      int row = half * 8 + r;
      pbuf[w][row * 32 + c]      = (__bf16)p0[r];
      pbuf[w][row * 32 + 16 + c] = (__bf16)p1[r];
    }
    asm volatile("s_wait_dscnt 0" ::: "memory");
    v16bf pA = load_A_bf16(&pbuf[w][0], 0, 0, 32, lane);

#pragma unroll
    for (int t = 0; t < 4; ++t) {
      v16bf vB = load_B_bf16(&vtb[cur][0], t * 16, 0, 32, lane);
      acc[t] = __builtin_amdgcn_wmma_f32_16x16x32_bf16(
          false, pA, false, vB, (short)0, acc[t], false, false);
    }

    wait_async0();
    __syncthreads();
  }

  float rinv[8];
#pragma unroll
  for (int r = 0; r < 8; ++r) rinv[r] = 1.0f / l8[r];
  int b = bh >> 4, h = bh & 15;
#pragma unroll
  for (int t = 0; t < 4; ++t) {
    int d = t * 16 + c;
#pragma unroll
    for (int r = 0; r < 8; ++r) {
      int row = q0 + half * 8 + r;
      oh[((size_t)b * T_SEQ + row) * C_DIM + h * HD + d] =
          (__bf16)(acc[t][r] * rinv[r]);
    }
  }
}

__global__ __launch_bounds__(256) void oproj_kernel(
    const __bf16* __restrict__ oh, const __bf16* __restrict__ Wo,
    const float* __restrict__ bo, float* __restrict__ out) {
  int lane = threadIdx.x & 31;
  int wid  = blockIdx.x * 8 + (threadIdx.x >> 5);
  int m0   = (wid >> 4) * 16;
  int n0   = (wid & 15) * 64;
  int half = lane >> 4, c = lane & 15;

  v8f acc[4];
#pragma unroll
  for (int t = 0; t < 4; ++t)
#pragma unroll
    for (int r = 0; r < 8; ++r) acc[t][r] = 0.f;

  for (int k0 = 0; k0 < C_DIM; k0 += 32) {
    v16bf a = load_A_bf16(oh, m0, k0, C_DIM, lane);
#pragma unroll
    for (int t = 0; t < 4; ++t) {
      v16bf b = load_B_bf16(Wo, n0 + t * 16, k0, C_DIM, lane);
      acc[t] = __builtin_amdgcn_wmma_f32_16x16x32_bf16(
          false, a, false, b, (short)0, acc[t], false, false);
    }
  }

#pragma unroll
  for (int t = 0; t < 4; ++t) {
    int n = n0 + t * 16 + c;
    float bn = bo[n];
#pragma unroll
    for (int r = 0; r < 8; ++r) {
      int m = m0 + half * 8 + r;
      out[(size_t)m * C_DIM + n] = acc[t][r] + bn;
    }
  }
}

extern "C" void kernel_launch(void* const* d_in, const int* in_sizes, int n_in,
                              void* d_out, int out_size, void* d_ws,
                              size_t ws_size, hipStream_t stream) {
  const float* x  = (const float*)d_in[0];
  const float* Wq = (const float*)d_in[1];
  const float* bq = (const float*)d_in[2];
  const float* Wk = (const float*)d_in[3];
  const float* bk = (const float*)d_in[4];
  const float* Wv = (const float*)d_in[5];
  const float* bv = (const float*)d_in[6];
  const float* Wo = (const float*)d_in[7];
  const float* bo = (const float*)d_in[8];
  float* out = (float*)d_out;

  const size_t MB = (size_t)1 << 20;
  char* ws = (char*)d_ws;
  __bf16* xb  = (__bf16*)(ws + 0);
  __bf16* Wqb = (__bf16*)(ws + 8  * MB);
  __bf16* Wkb = (__bf16*)(ws + 10 * MB);
  __bf16* Wvb = (__bf16*)(ws + 12 * MB);
  __bf16* Wob = (__bf16*)(ws + 14 * MB);
  signed char* qs = (signed char*)(ws + 16 * MB);
  signed char* ks = (signed char*)(ws + 20 * MB);
  __bf16* vT = (__bf16*)(ws + 24 * MB);
  __bf16* oh = (__bf16*)(ws + 32 * MB);

  {
    int nx = BT * C_DIM;
    cvt_bf16_kernel<<<(nx + 255) / 256, 256, 0, stream>>>(x, xb, nx);
    int nw = C_DIM * C_DIM;
    cvt_bf16_kernel<<<(nw + 255) / 256, 256, 0, stream>>>(Wq, Wqb, nw);
    cvt_bf16_kernel<<<(nw + 255) / 256, 256, 0, stream>>>(Wk, Wkb, nw);
    cvt_bf16_kernel<<<(nw + 255) / 256, 256, 0, stream>>>(Wv, Wvb, nw);
    cvt_bf16_kernel<<<(nw + 255) / 256, 256, 0, stream>>>(Wo, Wob, nw);
  }

  qkv_kernel<<<12288 / 8, 256, 0, stream>>>(xb, Wqb, Wkb, Wvb, bq, bk, bv,
                                            qs, ks, vT);

  attn_kernel<<<4096 / 8, 256, 0, stream>>>(qs, ks, vT, oh);

  oproj_kernel<<<4096 / 8, 256, 0, stream>>>(oh, Wob, bo, out);
}
